// RDESIRouter_25348896981064
// MI455X (gfx1250) — matrix operand order repaired and checked by reading, not compiled, not run on hardware
//
#include <hip/hip_runtime.h>
#include <hip/hip_bf16.h>

typedef __attribute__((ext_vector_type(16))) __bf16 v16bf;
typedef __attribute__((ext_vector_type(8)))  float  v8f;

#define HDIM 2048
#define EXPERTS 64
#define TOKENS 16384
#define ROWS_PER_BLOCK 128
#define ROWS_PER_WAVE 32          // two 16-row WMMA tiles per wave
#define BLOCK_THREADS 128         // 4 wave32

// ---------------------------------------------------------------------------
// Kernel 0: zero the 128-float global accumulator region in workspace
// ---------------------------------------------------------------------------
__global__ void rdesi_zero_ws(float* __restrict__ acc) {
    if (threadIdx.x < 128) acc[threadIdx.x] = 0.0f;
}

// ---------------------------------------------------------------------------
// Kernel 1: convert gate_w (fp32 [E,H]) to bf16, L2-resident (256 KB)
// ---------------------------------------------------------------------------
__global__ void rdesi_cvt_gate(const float* __restrict__ g,
                               __bf16* __restrict__ gb, int n) {
    int i = blockIdx.x * blockDim.x + threadIdx.x;
    if (i < n) gb[i] = (__bf16)g[i];
}

// ---------------------------------------------------------------------------
// A-fragment loader: contiguous-k A layout, k = 16h + s (lane = row M, K halves split by
// lane-half). Loads 8+8 contiguous fp32, packs to bf16 in-register.
// ---------------------------------------------------------------------------
__device__ __forceinline__ v16bf load_a_frag(const float* __restrict__ p) {
    const float4* q = reinterpret_cast<const float4*>(p);
    float4 a0 = q[0], a1 = q[1];   // K = koff .. koff+7
    float4 a2 = q[2], a3 = q[3];   // K = koff+8 .. koff+15
    v16bf A;
    A[0]  = (__bf16)a0.x; A[1]  = (__bf16)a0.y; A[2]  = (__bf16)a0.z; A[3]  = (__bf16)a0.w;
    A[4]  = (__bf16)a1.x; A[5]  = (__bf16)a1.y; A[6]  = (__bf16)a1.z; A[7]  = (__bf16)a1.w;
    A[8]  = (__bf16)a2.x; A[9]  = (__bf16)a2.y; A[10] = (__bf16)a2.z; A[11] = (__bf16)a2.w;
    A[12] = (__bf16)a3.x; A[13] = (__bf16)a3.y; A[14] = (__bf16)a3.z; A[15] = (__bf16)a3.w;
    return A;
}

// ---------------------------------------------------------------------------
// Kernel 2: fused router — bf16 WMMA GEMM + bias + top2 + softmax + aux parts
// 128 blocks x 128 threads (4 wave32). Each wave: 32 tokens x 64 experts
// (2 M-tiles x 4 N-tiles = 8 accumulators; each B fragment feeds 2 WMMAs).
// ---------------------------------------------------------------------------
__global__ __launch_bounds__(BLOCK_THREADS)
void rdesi_router_main(const float* __restrict__ x,
                       const __bf16* __restrict__ gateb,
                       const float* __restrict__ rep,
                       const float* __restrict__ loads,
                       const float* __restrict__ counts,
                       const int* __restrict__ td,
                       float* __restrict__ out,
                       float* __restrict__ gprob,
                       float* __restrict__ gcnt) {
    __shared__ float sm_scores[ROWS_PER_BLOCK * EXPERTS];   // 32 KB
    __shared__ float sm_tmax[ROWS_PER_BLOCK];
    __shared__ float sm_tsum[ROWS_PER_BLOCK];
    __shared__ float sm_prob[EXPERTS];
    __shared__ float sm_cnt[EXPERTS];

    const int tid   = threadIdx.x;
    const int wave  = tid >> 5;
    const int lane  = tid & 31;
    const int lhalf = lane >> 4;     // 0: lanes 0-15, 1: lanes 16-31
    const int lmod  = lane & 15;

    if (tid < EXPERTS) { sm_prob[tid] = 0.0f; sm_cnt[tid] = 0.0f; }

    const int rowBase = blockIdx.x * ROWS_PER_BLOCK + wave * ROWS_PER_WAVE;
    const int koffA = lhalf * 16;    // A: K-offset per lane-half
    const int koffB = lhalf * 16;    // B: K-offset per lane-half
    const float* xrow0 = x + (size_t)(rowBase      + lmod) * HDIM + koffA;
    const float* xrow1 = x + (size_t)(rowBase + 16 + lmod) * HDIM + koffA;

    const __bf16* gB[4];
    #pragma unroll
    for (int t = 0; t < 4; ++t)
        gB[t] = gateb + (size_t)(t * 16 + lmod) * HDIM + koffB;

    v8f acc[8];
    #pragma unroll
    for (int i = 0; i < 8; ++i) acc[i] = (v8f){};

    for (int k0 = 0; k0 < HDIM; k0 += 32) {
        // WGP-scope speculative prefetch ~4 iterations ahead; OOB addresses
        // past the tensor end are silently dropped (ISA 10.5).
        __builtin_prefetch(xrow0 + k0 + 128, 0, 3);
        __builtin_prefetch(xrow1 + k0 + 128, 0, 3);

        v16bf A0 = load_a_frag(xrow0 + k0);
        v16bf A1 = load_a_frag(xrow1 + k0);

        v16bf Bf[4];
        #pragma unroll
        for (int t = 0; t < 4; ++t)
            Bf[t] = *reinterpret_cast<const v16bf*>(gB[t] + k0);

        #pragma unroll
        for (int t = 0; t < 4; ++t) {
            acc[t]     = __builtin_amdgcn_wmma_f32_16x16x32_bf16(
                             false, A0, false, Bf[t], (short)0, acc[t],     false, false);
            acc[4 + t] = __builtin_amdgcn_wmma_f32_16x16x32_bf16(
                             false, A1, false, Bf[t], (short)0, acc[4 + t], false, false);
        }
    }

    // ---- epilogue: add per-expert bias, scatter to LDS per C-layout
    // C layout: lanes 0-15 -> M=j, lanes 16-31 -> M=8+j; N = lmod + 16*t
    const float logT = logf((float)td[0] + 1.0f);
    #pragma unroll
    for (int t = 0; t < 4; ++t) {
        const int e = t * 16 + lmod;
        const float bias = 0.1f * rep[e] - 0.1f * loads[e]
                         + 0.1f * sqrtf(logT / (counts[e] + 1e-10f));
        #pragma unroll
        for (int mt = 0; mt < 2; ++mt) {
            const v8f a = acc[mt * 4 + t];
            #pragma unroll
            for (int j = 0; j < 8; ++j) {
                const int r = wave * ROWS_PER_WAVE + mt * 16 + lhalf * 8 + j;
                sm_scores[r * EXPERTS + e] = a[j] + bias;
            }
        }
    }
    __syncthreads();

    // ---- phase 2a: per-token top-2, 2-way softmax, outputs, count atomics
    {
        const float* s = &sm_scores[tid * EXPERTS];
        float v0 = -INFINITY, v1 = -INFINITY;
        int i0 = 0, i1 = 0;
        for (int e = 0; e < EXPERTS; ++e) {
            const float se = s[e];
            if (se > v0)      { v1 = v0; i1 = i0; v0 = se; i0 = e; }
            else if (se > v1) { v1 = se; i1 = e; }
        }
        float sum = 0.0f;
        for (int e = 0; e < EXPERTS; ++e) sum += expf(s[e] - v0);
        sm_tmax[tid] = v0;
        sm_tsum[tid] = sum;

        const int tok = blockIdx.x * ROWS_PER_BLOCK + tid;
        const float e1  = expf(v1 - v0);
        const float inv = 1.0f / (1.0f + e1);
        out[tok * 2 + 0] = inv;                       // routing_weights
        out[tok * 2 + 1] = e1 * inv;
        out[2 * TOKENS + tok * 2 + 0] = (float)i0;    // expert_indices
        out[2 * TOKENS + tok * 2 + 1] = (float)i1;
        atomicAdd(&sm_cnt[i0], 1.0f);
        atomicAdd(&sm_cnt[i1], 1.0f);
    }
    __syncthreads();

    // ---- phase 2b: per-expert router-prob partial sums (ds_add_f32)
    {
        const int e  = tid & 63;
        const int t0 = tid >> 6;  // 2 partial lanes per expert
        float accp = 0.0f;
        for (int t = t0; t < ROWS_PER_BLOCK; t += 2)
            accp += expf(sm_scores[t * EXPERTS + e] - sm_tmax[t]) / sm_tsum[t];
        atomicAdd(&sm_prob[e], accp);
    }
    __syncthreads();

    // ---- flush block partials to global accumulators
    if (tid < EXPERTS) {
        atomicAdd(&gprob[tid], sm_prob[tid]);
        atomicAdd(&gcnt[tid],  sm_cnt[tid]);
    }
}

// ---------------------------------------------------------------------------
// Kernel 3: finalize aux loss
// ---------------------------------------------------------------------------
__global__ void rdesi_finalize(const float* __restrict__ gprob,
                               const float* __restrict__ gcnt,
                               float* __restrict__ out) {
    __shared__ float red[EXPERTS];
    const int e = threadIdx.x;
    const float tpe  = gcnt[e]  * (1.0f / (float)TOKENS);
    const float rppe = gprob[e] * (1.0f / (float)TOKENS);
    red[e] = tpe * rppe * (float)EXPERTS;
    __syncthreads();
    if (e == 0) {
        float s = 0.0f;
        for (int i = 0; i < EXPERTS; ++i) s += red[i];
        out[4 * TOKENS] = s;   // aux_loss at offset 65536
    }
}

// ---------------------------------------------------------------------------
extern "C" void kernel_launch(void* const* d_in, const int* in_sizes, int n_in,
                              void* d_out, int out_size, void* d_ws, size_t ws_size,
                              hipStream_t stream) {
    const float* x      = (const float*)d_in[0];
    const float* gate_w = (const float*)d_in[1];
    const float* rep    = (const float*)d_in[2];
    const float* loads  = (const float*)d_in[3];
    const float* counts = (const float*)d_in[4];
    const int*   td     = (const int*)d_in[5];
    float* out = (float*)d_out;

    float*  gprob = (float*)d_ws;          // [64]
    float*  gcnt  = gprob + 64;            // [64]
    __bf16* gateb = (__bf16*)((char*)d_ws + 1024);  // [E*H] bf16, 256 KB

    hipLaunchKernelGGL(rdesi_zero_ws, dim3(1), dim3(128), 0, stream, gprob);
    hipLaunchKernelGGL(rdesi_cvt_gate, dim3((EXPERTS * HDIM) / 256), dim3(256),
                       0, stream, gate_w, gateb, EXPERTS * HDIM);
    hipLaunchKernelGGL(rdesi_router_main, dim3(TOKENS / ROWS_PER_BLOCK),
                       dim3(BLOCK_THREADS), 0, stream,
                       x, gateb, rep, loads, counts, td, out, gprob, gcnt);
    hipLaunchKernelGGL(rdesi_finalize, dim3(1), dim3(64), 0, stream, gprob, gcnt, out);
}
